// BiLSTM_39745627357363
// MI455X (gfx1250) — matrix operand order repaired and checked by reading, not compiled, not run on hardware
//
#include <hip/hip_runtime.h>

// ---------------------------------------------------------------------------
// Problem constants (match reference)
// ---------------------------------------------------------------------------
constexpr int Bq = 32;     // batch
constexpr int Sq = 512;    // seq len
constexpr int Eq = 128;    // embed dim
constexpr int Hq = 256;    // hidden
constexpr int Lq = 32;     // tags
constexpr int NG = 3 * Hq; // 768 gate width
constexpr int Mtot = Bq * Sq; // 16384 rows

// ---------------------------------------------------------------------------
// WMMA types / helpers (gfx1250, wave32)
// ---------------------------------------------------------------------------
typedef __bf16  v16bf __attribute__((ext_vector_type(16)));
typedef float   v8f   __attribute__((ext_vector_type(8)));

union Frag16 {           // 16 bf16 = 8 VGPRs
    uint4 q[2];
    v16bf v;
};

static __device__ __forceinline__ unsigned short f2bf(float f) {
    unsigned u = __float_as_uint(f);
    u += 0x7FFFu + ((u >> 16) & 1u);        // round-to-nearest-even
    return (unsigned short)(u >> 16);
}
static __device__ __forceinline__ float bf2f(unsigned short h) {
    return __uint_as_float(((unsigned)h) << 16);
}
static __device__ __forceinline__ float sigm(float x) {
    return 1.0f / (1.0f + __expf(-x));
}
static __device__ __forceinline__ v8f wmma_bf16(const Frag16& a, const Frag16& b, v8f c) {
    return __builtin_amdgcn_wmma_f32_16x16x32_bf16(false, a.v, false, b.v,
                                                   (short)0, c, false, false);
}

// ---------------------------------------------------------------------------
// K0: pack a row-major [K,N] f32 weight into per-tile bf16 B-fragments.
// Fragment order: tile = kt*(N/16)+nt; within tile, lane l holds col n=l&15,
// k = kt*32 + ((l&16)?8:0) + j + (j>=8?8:0)  for j=0..15, stored contiguously.
// ---------------------------------------------------------------------------
__global__ void pack_weight_kernel(const float* __restrict__ src,
                                   unsigned short* __restrict__ dst,
                                   int K, int N) {
    int tile = blockIdx.x;
    int ntiles = N >> 4;
    int kt = tile / ntiles, nt = tile % ntiles;
    int l = threadIdx.x;
    int n = nt * 16 + (l & 15);
    int kOff = (l & 16) ? 8 : 0;
    unsigned short* o = dst + ((size_t)tile * 32 + l) * 16;
#pragma unroll
    for (int j = 0; j < 16; ++j) {
        int k = kt * 32 + kOff + j + ((j >= 8) ? 8 : 0);
        o[j] = f2bf(src[(size_t)k * N + n]);
    }
}

// ---------------------------------------------------------------------------
// K1: gather embedding rows and cast to bf16.  ids[row*idStride] selects row.
// ---------------------------------------------------------------------------
__global__ void gather_cast_kernel(const int* __restrict__ ids, int idStride,
                                   const float* __restrict__ emb,
                                   unsigned short* __restrict__ dst) {
    int row = blockIdx.x;
    int e = threadIdx.x;
    int id = ids[(size_t)row * idStride];
    dst[(size_t)row * Eq + e] = f2bf(emb[(size_t)id * Eq + e]);
}

__global__ void zero_i32_kernel(int* __restrict__ p, int n) {
    int i = blockIdx.x * blockDim.x + threadIdx.x;
    if (i < n) p[i] = 0;
}

// Build skip_rev: rev_pos = S - t - wlen for valid words.
__global__ void skiprev_kernel(const int* __restrict__ skip,
                               int* __restrict__ skip_rev) {
    int idx = blockIdx.x * blockDim.x + threadIdx.x;
    if (idx >= Bq * Sq) return;
    int b = idx / Sq, t = idx % Sq;
    int wl = skip[(size_t)idx * 2 + 1];
    if (wl > 0) {
        int rp = Sq - t - wl;
        if (rp >= 0 && rp < Sq) {
            skip_rev[((size_t)b * Sq + rp) * 2 + 0] = skip[(size_t)idx * 2 + 0];
            skip_rev[((size_t)b * Sq + rp) * 2 + 1] = wl;
        }
    }
}

// ---------------------------------------------------------------------------
// K2: input-projection GEMM  Out[M,N](bf16) = A[M,K](bf16) @ Wpk + bias.
// One 16x16 C-tile per wave; 8 waves / block.  rev=1 remaps A row b*S+t ->
// b*S+(S-1-t) (for the backward direction's reversed inputs).
// ---------------------------------------------------------------------------
__global__ __launch_bounds__(256)
void proj_gemm_kernel(const unsigned short* __restrict__ A, int rev,
                      const unsigned short* __restrict__ Bpk,
                      const float* __restrict__ bias,
                      unsigned short* __restrict__ Out,
                      int N, int K) {
    int ntiles = N >> 4;
    int wave = threadIdx.x >> 5;
    int lane = threadIdx.x & 31;
    int tile = blockIdx.x * 8 + wave;
    int mt = tile / ntiles, nt = tile % ntiles;

    int n = nt * 16 + (lane & 15);
    int hi8 = (lane & 16) ? 8 : 0;
    int kA0 = (lane & 16) ? 8 : 0;

    v8f acc;
#pragma unroll
    for (int v = 0; v < 8; ++v) acc[v] = bias[n];

    int arow = mt * 16 + (lane & 15);
    if (rev) {
        int b = arow >> 9, t = arow & (Sq - 1);
        arow = (b << 9) + (Sq - 1 - t);
    }
    const int kslices = K >> 5;
    for (int kt = 0; kt < kslices; ++kt) {
        Frag16 fa, fb;
        const unsigned short* ap = A + (size_t)arow * K + kt * 32 + kA0;
        fa.q[0] = *(const uint4*)(ap);
        fa.q[1] = *(const uint4*)(ap + 16);
        const unsigned short* bp = Bpk + ((size_t)(kt * ntiles + nt) * 32 + lane) * 16;
        fb.q[0] = *(const uint4*)(bp);
        fb.q[1] = *(const uint4*)(bp + 8);
        acc = wmma_bf16(fa, fb, acc);
    }
#pragma unroll
    for (int v = 0; v < 8; ++v) {
        int m = mt * 16 + hi8 + v;
        Out[(size_t)m * N + n] = f2bf(acc[v]);
    }
}

// ---------------------------------------------------------------------------
// K3: lattice LSTM scan.  One 512-thread (16-wave) block per direction.
// State in LDS; per step:   pre = preX[t] + h@Whh ; apre = aX[t] + pc@aWhh
//   -> gates/merge ->  pw = preW[t] + h_new@wWhh  -> word-cell update.
// ---------------------------------------------------------------------------
struct ScanArgs {
    const unsigned short* preX;   // [B*S, 768] bf16 (includes bias)
    const unsigned short* aX;     // [B*S, 256] bf16 (includes ab)
    const unsigned short* preW;   // [B*S, 768] bf16 (includes wb)
    const unsigned short* Whh;    // packed bf16, Ntiles=48
    const unsigned short* aWhh;   // packed bf16, Ntiles=16
    const unsigned short* wWhh;   // packed bf16, Ntiles=48
    const int*            skip;   // [B,S,2] in this direction's time order
    unsigned short*       hout;   // [B,S,256] bf16, forward time order
    int                   rev;    // 1 => store h at S-1-t
};

// dynamic-LDS layout (bytes)
constexpr size_t SM_H    = 0;                       // ushort[32][256] 16 KB
constexpr size_t SM_PCB  = SM_H    + 16384;         // ushort[32][256] 16 KB
constexpr size_t SM_C    = SM_PCB  + 16384;         // float [32][256] 32 KB
constexpr size_t SM_PC   = SM_C    + 32768;         // float [32][256] 32 KB
constexpr size_t SM_PRE  = SM_PC   + 32768;         // float [32][768] 96 KB
constexpr size_t SM_APRE = SM_PRE  + 98304;         // float [32][256] 32 KB
constexpr size_t SM_SCAL = SM_APRE + 32768;         // int   [4][32]   0.5 KB
constexpr size_t SM_TOTAL = SM_SCAL + 512;          // ~225 KB of 320 KB WGP LDS

__global__ __launch_bounds__(512)
void lattice_scan_kernel(ScanArgs a0, ScanArgs a1) {
    extern __shared__ char smem[];
    const ScanArgs a = (blockIdx.x == 0) ? a0 : a1;

    unsigned short* sh_h   = (unsigned short*)(smem + SM_H);
    unsigned short* sh_pcb = (unsigned short*)(smem + SM_PCB);
    float*          sh_c   = (float*)(smem + SM_C);
    float*          sh_pc  = (float*)(smem + SM_PC);
    float*          sh_pre = (float*)(smem + SM_PRE);
    float*          sh_ap  = (float*)(smem + SM_APRE);
    int*            sh_pcnt = (int*)(smem + SM_SCAL);
    int*            sh_pval = sh_pcnt + 32;
    int*            sh_hasw = sh_pval + 32;
    int*            sh_wl   = sh_hasw + 32;

    const int tid  = threadIdx.x;
    const int wave = tid >> 5;
    const int lane = tid & 31;

    for (int p = tid; p < Bq * Hq; p += 512) {
        sh_h[p] = 0; sh_pcb[p] = 0;
        sh_c[p] = 0.0f; sh_pc[p] = 0.0f;
    }
    if (tid < 32) { sh_pcnt[tid] = -1; sh_pval[tid] = 0; }
    __syncthreads();

    const int hi8 = (lane & 16) ? 8 : 0;
    const int kA0 = (lane & 16) ? 8 : 0;
    const int nlo = lane & 15;

    for (int t = 0; t < Sq; ++t) {
        // ---- Phase A: pre = preX + h@Whh (96 tiles), apre = aX + pc@aWhh (32)
        for (int it = wave; it < 128; it += 16) {
            const bool isMain = it < 96;
            const int  ntiles = isMain ? 48 : 16;
            const int  Ncols  = isMain ? NG : Hq;
            const int  lt     = isMain ? it : it - 96;
            const int  mt     = lt / ntiles, nt = lt % ntiles;
            const unsigned short* Cini = isMain ? a.preX : a.aX;
            const unsigned short* Bpk  = isMain ? a.Whh  : a.aWhh;
            const unsigned short* Asrc = isMain ? sh_h   : sh_pcb;
            float*                Odst = isMain ? sh_pre : sh_ap;

            const int n = nt * 16 + nlo;
            v8f acc;
#pragma unroll
            for (int v = 0; v < 8; ++v) {
                int b = mt * 16 + hi8 + v;
                acc[v] = bf2f(Cini[((size_t)b * Sq + t) * Ncols + n]);
            }
            const int mrow = mt * 16 + nlo;
#pragma unroll
            for (int kt = 0; kt < 8; ++kt) {
                Frag16 fa, fb;
                const unsigned short* ap = Asrc + mrow * Hq + kt * 32 + kA0;
                fa.q[0] = *(const uint4*)(ap);
                fa.q[1] = *(const uint4*)(ap + 16);
                const unsigned short* bp = Bpk + ((size_t)(kt * ntiles + nt) * 32 + lane) * 16;
                fb.q[0] = *(const uint4*)(bp);
                fb.q[1] = *(const uint4*)(bp + 8);
                acc = wmma_bf16(fa, fb, acc);
            }
#pragma unroll
            for (int v = 0; v < 8; ++v) {
                int m = mt * 16 + hi8 + v;
                Odst[m * Ncols + n] = acc[v];
            }
        }
        __syncthreads();

        // ---- Phase B: gates, merge, h_new
        for (int p = tid; p < Bq * Hq; p += 512) {
            int b = p >> 8, j = p & 255;
            float i  = sigm(sh_pre[b * NG + j]);
            float o  = sigm(sh_pre[b * NG + Hq + j]);
            float g  = tanhf(sh_pre[b * NG + 2 * Hq + j]);
            float al = sigm(sh_ap[b * Hq + j]);
            bool  mg = (sh_pval[b] != 0) && (sh_pcnt[b] == 0);
            float c  = sh_c[p];
            float pc = sh_pc[p];
            float c_pl = (1.0f - i) * c + i * g;
            float ei = __expf(i), ea = __expf(al);
            float c_mg = (ei * g + ea * pc) / (ei + ea);
            float cn = mg ? c_mg : c_pl;
            float hn = o * tanhf(cn);
            sh_c[p] = cn;
            sh_h[p] = f2bf(hn);
            int ot = a.rev ? (Sq - 1 - t) : t;
            a.hout[((size_t)b * Sq + ot) * Hq + j] = f2bf(hn);
        }
        __syncthreads();

        // ---- Phase C: scalar state + pw = preW + h_new@wWhh (96 tiles)
        if (tid < 32) {
            int b = tid;
            bool mg = (sh_pval[b] != 0) && (sh_pcnt[b] == 0);
            sh_pval[b] = (sh_pval[b] != 0) && !mg;
            sh_pcnt[b] = sh_pcnt[b] - 1;
            int wl = a.skip[((size_t)b * Sq + t) * 2 + 1];
            sh_wl[b] = wl;
            sh_hasw[b] = (wl >= 2);
        }
        for (int it = wave; it < 96; it += 16) {
            const int mt = it / 48, nt = it % 48;
            const int n = nt * 16 + nlo;
            v8f acc;
#pragma unroll
            for (int v = 0; v < 8; ++v) {
                int b = mt * 16 + hi8 + v;
                acc[v] = bf2f(a.preW[((size_t)b * Sq + t) * NG + n]);
            }
            const int mrow = mt * 16 + nlo;
#pragma unroll
            for (int kt = 0; kt < 8; ++kt) {
                Frag16 fa, fb;
                const unsigned short* ap = sh_h + mrow * Hq + kt * 32 + kA0;
                fa.q[0] = *(const uint4*)(ap);
                fa.q[1] = *(const uint4*)(ap + 16);
                const unsigned short* bp = a.wWhh + ((size_t)(kt * 48 + nt) * 32 + lane) * 16;
                fb.q[0] = *(const uint4*)(bp);
                fb.q[1] = *(const uint4*)(bp + 8);
                acc = wmma_bf16(fa, fb, acc);
            }
#pragma unroll
            for (int v = 0; v < 8; ++v) {
                int m = mt * 16 + hi8 + v;
                sh_pre[m * NG + n] = acc[v];
            }
        }
        __syncthreads();

        // ---- Phase D: word cell
        for (int p = tid; p < Bq * Hq; p += 512) {
            int b = p >> 8, j = p & 255;
            if (sh_hasw[b]) {
                float iw = sh_pre[b * NG + j];
                float fw = sh_pre[b * NG + Hq + j];
                float gw = sh_pre[b * NG + 2 * Hq + j];
                float cw = sigm(fw) * sh_c[p] + sigm(iw) * tanhf(gw);
                sh_pc[p]  = cw;
                sh_pcb[p] = f2bf(cw);
            }
        }
        if (tid < 32 && sh_hasw[tid]) {
            sh_pcnt[tid] = sh_wl[tid] - 1;
            sh_pval[tid] = 1;
        }
        __syncthreads();
    }
}

// ---------------------------------------------------------------------------
// K4: tag projection  out[M,32] = [h_fwd | h_bwd] @ tag_W + tag_b
// ---------------------------------------------------------------------------
__global__ __launch_bounds__(256)
void tag_gemm_kernel(const unsigned short* __restrict__ hf,
                     const unsigned short* __restrict__ hb,
                     const unsigned short* __restrict__ Wpk,  // K=512,N=32 packed
                     const float* __restrict__ tagb,
                     float* __restrict__ out) {
    int wave = threadIdx.x >> 5;
    int lane = threadIdx.x & 31;
    int tile = blockIdx.x * 8 + wave;          // 2048 tiles total
    int mt = tile >> 1, nt = tile & 1;
    int n = nt * 16 + (lane & 15);
    int hi8 = (lane & 16) ? 8 : 0;
    int kA0 = (lane & 16) ? 8 : 0;

    v8f acc;
#pragma unroll
    for (int v = 0; v < 8; ++v) acc[v] = tagb[n];

    int arow = mt * 16 + (lane & 15);
#pragma unroll
    for (int kt = 0; kt < 16; ++kt) {
        const unsigned short* Asrc = (kt < 8) ? hf : hb;
        int kk = (kt < 8) ? kt : (kt - 8);
        Frag16 fa, fb;
        const unsigned short* ap = Asrc + (size_t)arow * Hq + kk * 32 + kA0;
        fa.q[0] = *(const uint4*)(ap);
        fa.q[1] = *(const uint4*)(ap + 16);
        const unsigned short* bp = Wpk + ((size_t)(kt * 2 + nt) * 32 + lane) * 16;
        fb.q[0] = *(const uint4*)(bp);
        fb.q[1] = *(const uint4*)(bp + 8);
        acc = wmma_bf16(fa, fb, acc);
    }
#pragma unroll
    for (int v = 0; v < 8; ++v) {
        int m = mt * 16 + hi8 + v;
        out[(size_t)m * Lq + n] = acc[v];
    }
}

// ---------------------------------------------------------------------------
// Host side
// ---------------------------------------------------------------------------
extern "C" void kernel_launch(void* const* d_in, const int* in_sizes, int n_in,
                              void* d_out, int out_size, void* d_ws, size_t ws_size,
                              hipStream_t stream) {
    const int*   component_ids = (const int*)d_in[0];
    const int*   skip_input    = (const int*)d_in[1];
    const float* emb           = (const float*)d_in[2];
    const float* tag_W         = (const float*)d_in[3];
    const float* tag_b         = (const float*)d_in[4];
    const float* W[18];
    for (int i = 0; i < 18; ++i) W[i] = (const float*)d_in[5 + i];
    // per-direction name order: Wih, Whh, b, aWih, aWhh, ab, wWih, wWhh, wb
    const float *fw_Wih = W[0], *fw_Whh = W[1], *fw_b  = W[2],
                *fw_aWih = W[3], *fw_aWhh = W[4], *fw_ab = W[5],
                *fw_wWih = W[6], *fw_wWhh = W[7], *fw_wb = W[8];
    const float *bw_Wih = W[9], *bw_Whh = W[10], *bw_b  = W[11],
                *bw_aWih = W[12], *bw_aWhh = W[13], *bw_ab = W[14],
                *bw_wWih = W[15], *bw_wWhh = W[16], *bw_wb = W[17];

    float* outp = (float*)d_out;
    char*  ws   = (char*)d_ws;
    size_t off = 0;
    auto take = [&](size_t bytes) -> char* {
        char* p = ws + off;
        off += (bytes + 255) & ~(size_t)255;
        return p;
    };
    auto us = [](char* p) { return (unsigned short*)p; };

    unsigned short* Xbf = us(take((size_t)Mtot * Eq * 2));
    unsigned short* WEf = us(take((size_t)Mtot * Eq * 2));
    unsigned short* WEb = us(take((size_t)Mtot * Eq * 2));
    int* skiprev        = (int*)take((size_t)Bq * Sq * 2 * 4);

    unsigned short* pk_fw_Wih  = us(take((size_t)Eq * NG * 2));
    unsigned short* pk_fw_Whh  = us(take((size_t)Hq * NG * 2));
    unsigned short* pk_fw_aWih = us(take((size_t)Eq * Hq * 2));
    unsigned short* pk_fw_aWhh = us(take((size_t)Hq * Hq * 2));
    unsigned short* pk_fw_wWih = us(take((size_t)Eq * NG * 2));
    unsigned short* pk_fw_wWhh = us(take((size_t)Hq * NG * 2));
    unsigned short* pk_bw_Wih  = us(take((size_t)Eq * NG * 2));
    unsigned short* pk_bw_Whh  = us(take((size_t)Hq * NG * 2));
    unsigned short* pk_bw_aWih = us(take((size_t)Eq * Hq * 2));
    unsigned short* pk_bw_aWhh = us(take((size_t)Hq * Hq * 2));
    unsigned short* pk_bw_wWih = us(take((size_t)Eq * NG * 2));
    unsigned short* pk_bw_wWhh = us(take((size_t)Hq * NG * 2));
    unsigned short* pk_tagW    = us(take((size_t)(2 * Hq) * Lq * 2));

    unsigned short* preXf = us(take((size_t)Mtot * NG * 2));
    unsigned short* aXf   = us(take((size_t)Mtot * Hq * 2));
    unsigned short* preWf = us(take((size_t)Mtot * NG * 2));
    unsigned short* preXb = us(take((size_t)Mtot * NG * 2));
    unsigned short* aXb   = us(take((size_t)Mtot * Hq * 2));
    unsigned short* preWb = us(take((size_t)Mtot * NG * 2));
    unsigned short* hfbf  = us(take((size_t)Mtot * Hq * 2));
    unsigned short* hbbf  = us(take((size_t)Mtot * Hq * 2));

    auto pack = [&](const float* src, unsigned short* dst, int K, int N) {
        pack_weight_kernel<<<(K / 32) * (N / 16), 32, 0, stream>>>(src, dst, K, N);
    };
    pack(fw_Wih,  pk_fw_Wih,  Eq, NG);
    pack(fw_Whh,  pk_fw_Whh,  Hq, NG);
    pack(fw_aWih, pk_fw_aWih, Eq, Hq);
    pack(fw_aWhh, pk_fw_aWhh, Hq, Hq);
    pack(fw_wWih, pk_fw_wWih, Eq, NG);
    pack(fw_wWhh, pk_fw_wWhh, Hq, NG);
    pack(bw_Wih,  pk_bw_Wih,  Eq, NG);
    pack(bw_Whh,  pk_bw_Whh,  Hq, NG);
    pack(bw_aWih, pk_bw_aWih, Eq, Hq);
    pack(bw_aWhh, pk_bw_aWhh, Hq, Hq);
    pack(bw_wWih, pk_bw_wWih, Eq, NG);
    pack(bw_wWhh, pk_bw_wWhh, Hq, NG);
    pack(tag_W,   pk_tagW,    2 * Hq, Lq);

    gather_cast_kernel<<<Mtot, Eq, 0, stream>>>(component_ids, 1, emb, Xbf);
    gather_cast_kernel<<<Mtot, Eq, 0, stream>>>(skip_input,    2, emb, WEf);
    zero_i32_kernel<<<(Bq * Sq * 2 + 255) / 256, 256, 0, stream>>>(skiprev, Bq * Sq * 2);
    skiprev_kernel<<<(Bq * Sq + 255) / 256, 256, 0, stream>>>(skip_input, skiprev);
    gather_cast_kernel<<<Mtot, Eq, 0, stream>>>(skiprev, 2, emb, WEb);

    const int tilesNG = (Mtot / 16) * (NG / 16) / 8;  // 6144 blocks
    const int tilesH  = (Mtot / 16) * (Hq / 16) / 8;  // 2048 blocks
    proj_gemm_kernel<<<tilesNG, 256, 0, stream>>>(Xbf, 0, pk_fw_Wih,  fw_b,  preXf, NG, Eq);
    proj_gemm_kernel<<<tilesH,  256, 0, stream>>>(Xbf, 0, pk_fw_aWih, fw_ab, aXf,   Hq, Eq);
    proj_gemm_kernel<<<tilesNG, 256, 0, stream>>>(WEf, 0, pk_fw_wWih, fw_wb, preWf, NG, Eq);
    proj_gemm_kernel<<<tilesNG, 256, 0, stream>>>(Xbf, 1, pk_bw_Wih,  bw_b,  preXb, NG, Eq);
    proj_gemm_kernel<<<tilesH,  256, 0, stream>>>(Xbf, 1, pk_bw_aWih, bw_ab, aXb,   Hq, Eq);
    proj_gemm_kernel<<<tilesNG, 256, 0, stream>>>(WEb, 0, pk_bw_wWih, bw_wb, preWb, NG, Eq);

    ScanArgs afw{preXf, aXf, preWf, pk_fw_Whh, pk_fw_aWhh, pk_fw_wWhh,
                 skip_input, hfbf, 0};
    ScanArgs abw{preXb, aXb, preWb, pk_bw_Whh, pk_bw_aWhh, pk_bw_wWhh,
                 skiprev, hbbf, 1};
    lattice_scan_kernel<<<2, 512, SM_TOTAL, stream>>>(afw, abw);

    tag_gemm_kernel<<<(Mtot / 16) * (Lq / 16) / 8, 256, 0, stream>>>(
        hfbf, hbbf, pk_tagW, tag_b, outp);
}
